// RNN_51737176048394
// MI455X (gfx1250) — matrix operand order repaired and checked by reading, not compiled, not run on hardware
//
#include <hip/hip_runtime.h>
#include <hip/hip_bf16.h>
#include <math.h>

#define VOCAB   256
#define N_EMBD  512
#define NH      512
#define BATCH   256
#define TSEQ    1024
#define NCOMB   768     // NH (Wh) + VOCAB (lm_W) fused N dimension
#define BCHUNK  16      // batch rows per workgroup (one M tile)
#define NTILES  6       // 16x16 N-tiles per wave: 8 waves * 6 * 16 = 768

typedef __attribute__((ext_vector_type(16))) __bf16 v16bf;
typedef __attribute__((ext_vector_type(8)))  __bf16 v8bf;
typedef __attribute__((ext_vector_type(8)))  float  v8f;

union ABf { v16bf v; v8bf h[2]; };

// ---------------------------------------------------------------------------
// Prep 1: Eproj[v][h] = sum_d wte[v][d] * W_xh[d][h] + b_xh[h]   (256x512 f32)
// Collapses the 137-GFLOP "emb @ Wx" into a 512 KB lookup table.
// ---------------------------------------------------------------------------
__global__ __launch_bounds__(256)
void rnn_prep_eproj(const float* __restrict__ wte,
                    const float* __restrict__ W_xh,
                    const float* __restrict__ b_xh,
                    float* __restrict__ Eproj) {
    const int id = blockIdx.x * 256 + threadIdx.x;   // VOCAB*NH threads
    const int v  = id >> 9;                          // NH == 512
    const int h  = id & (NH - 1);
    float acc = b_xh[h];
    const float* wrow = wte + (size_t)v * N_EMBD;    // scalar (block-uniform) loads
    #pragma unroll 8
    for (int d = 0; d < N_EMBD; ++d)
        acc = fmaf(wrow[d], W_xh[(size_t)d * NH + h], acc);
    Eproj[id] = acc;
}

// ---------------------------------------------------------------------------
// Prep 2: Wcomb (512 x 768 bf16) = [ Wh | lm_W ] row-major, K-major rows.
// ---------------------------------------------------------------------------
__global__ __launch_bounds__(256)
void rnn_prep_wcomb(const float* __restrict__ W_xh,
                    const float* __restrict__ lm_W,
                    __bf16* __restrict__ Wcomb) {
    const int id = blockIdx.x * 256 + threadIdx.x;   // NH*NCOMB threads
    const int k  = id / NCOMB;
    const int n  = id - k * NCOMB;
    const float val = (n < NH) ? W_xh[(size_t)(N_EMBD + k) * NH + n]
                               : lm_W[(size_t)k * VOCAB + (n - NH)];
    Wcomb[id] = (__bf16)val;
}

// ---------------------------------------------------------------------------
// Fused recurrent scan + LM head.
// grid = 16 blocks (one per 16-row batch chunk), 256 threads = 8 waves.
// Per step t:   acc(16x768) = h(16x512) @ Wcomb(512x768)   [bf16 WMMA, f32 acc]
//   cols [0,512):  h_new = tanh(acc + Eproj[token])  -> LDS (double-buffered)
//   cols [512,768): logits = acc + lm_b              -> d_out
// One __syncthreads per step (double-buffered h and token staging).
// ---------------------------------------------------------------------------
__global__ __launch_bounds__(256)
void rnn_scan_fused(const int*   __restrict__ idx,
                    const float* __restrict__ start,
                    const float* __restrict__ lm_b,
                    const float* __restrict__ Eproj,
                    const __bf16* __restrict__ Wcomb,
                    float* __restrict__ out) {
    __shared__ __bf16 hbuf[2][BCHUNK * NH];   // 2 x 16 KB
    __shared__ int    stok[2][BCHUNK];

    const int tid  = threadIdx.x;
    const int wave = tid >> 5;
    const int lane = tid & 31;
    const int half = lane >> 4;     // lane group (K/M split in WMMA layouts)
    const int ncol = lane & 15;
    const int b0   = blockIdx.x * BCHUNK;

    // h0 = start row broadcast to all 16 batch rows
    for (int e = tid; e < BCHUNK * NH; e += 256)
        hbuf[0][e] = (__bf16)start[e & (NH - 1)];

    for (int t = 0; t < TSEQ; ++t) {
        const int rb = t & 1, wb = rb ^ 1;
        if (tid < BCHUNK)
            stok[rb][tid] = idx[(size_t)(b0 + tid) * TSEQ + t];
        __syncthreads();   // h_{t-1} (prev epilogue) + this step's tokens visible

        v8f acc[NTILES] = {};
        const __bf16* hb = hbuf[rb];
        const int m_a = ncol;                  // A frag: lanes 0-15 / 16-31 -> M = lane&15

        #pragma unroll 4
        for (int kk = 0; kk < 16; ++kk) {      // K = 512 in steps of 32
            const int kbase = kk * 32;
            // A 16x32 bf16: VGPR0-3 = K[kbase + half*8 ..+7], VGPR4-7 = +16
            ABf a;
            a.h[0] = *(const v8bf*)&hb[m_a * NH + kbase + half * 8];
            a.h[1] = *(const v8bf*)&hb[m_a * NH + kbase + 16 + half * 8];
            // B 32x16 bf16: lane l holds column N = n0 + (l&15), K = kbase + half*8 + s (s<8), kbase + 16 + half*8 + (s-8) (s>=8)
            const __bf16* brow = Wcomb + (size_t)(kbase + half * 8) * NCOMB + wave * (NTILES * 16) + ncol;
            #pragma unroll
            for (int i = 0; i < NTILES; ++i) {
                v16bf b;
                #pragma unroll
                for (int s = 0; s < 8; ++s) {
                    b[s]     = brow[(size_t)s * NCOMB + i * 16];
                    b[s + 8] = brow[(size_t)(16 + s) * NCOMB + i * 16];
                }
                acc[i] = __builtin_amdgcn_wmma_f32_16x16x32_bf16(
                    false, a.v, false, b, (short)0, acc[i], false, false);
            }
        }

        // Epilogue. C/D layout: VGPR r, lane -> M = r + 8*half, N = n0 + (lane&15)
        #pragma unroll
        for (int i = 0; i < NTILES; ++i) {
            const int n0 = wave * (NTILES * 16) + i * 16;
            if (n0 < NH) {                     // hidden tile -> tanh -> LDS (write buffer)
                #pragma unroll
                for (int r = 0; r < 8; ++r) {
                    const int m   = r + half * 8;
                    const int tok = stok[rb][m];
                    const float xp = Eproj[tok * NH + n0 + ncol];
                    hbuf[wb][m * NH + n0 + ncol] = (__bf16)tanhf(acc[i][r] + xp);
                }
            } else {                           // logits tile -> global (coalesced per row)
                const int j  = n0 - NH + ncol;
                const float lb = lm_b[j];
                #pragma unroll
                for (int r = 0; r < 8; ++r) {
                    const int m = r + half * 8;
                    out[((size_t)(b0 + m) * TSEQ + t) * VOCAB + j] = acc[i][r] + lb;
                }
            }
        }
        // no second barrier: writes go to hbuf[wb]/out while reads were from hbuf[rb]
    }
}

// ---------------------------------------------------------------------------
extern "C" void kernel_launch(void* const* d_in, const int* in_sizes, int n_in,
                              void* d_out, int out_size, void* d_ws, size_t ws_size,
                              hipStream_t stream) {
    const int*   idx   = (const int*)  d_in[0];
    const float* wte   = (const float*)d_in[1];
    const float* start = (const float*)d_in[2];
    const float* W_xh  = (const float*)d_in[3];
    const float* b_xh  = (const float*)d_in[4];
    const float* lm_W  = (const float*)d_in[5];
    const float* lm_b  = (const float*)d_in[6];
    float* out = (float*)d_out;

    // workspace: [0, 512KB) Eproj f32 (256x512); [512KB, 1.25MB) Wcomb bf16 (512x768)
    float*  Eproj = (float*)d_ws;
    __bf16* Wcomb = (__bf16*)((char*)d_ws + (512 << 10));

    rnn_prep_eproj<<<(VOCAB * NH) / 256, 256, 0, stream>>>(wte, W_xh, b_xh, Eproj);
    rnn_prep_wcomb<<<(NH * NCOMB) / 256, 256, 0, stream>>>(W_xh, lm_W, Wcomb);
    rnn_scan_fused<<<BATCH / BCHUNK, 256, 0, stream>>>(idx, start, lm_b,
                                                       Eproj, Wcomb, out);
}
